// LinearCausalSelfAttention_65326452572771
// MI455X (gfx1250) — matrix operand order repaired and checked by reading, not compiled, not run on hardware
//
#include <hip/hip_runtime.h>
#include <hip/hip_bf16.h>

typedef __attribute__((ext_vector_type(16))) __bf16    v16bf;
typedef __attribute__((ext_vector_type(8)))  float     v8f;
typedef __attribute__((ext_vector_type(4)))  unsigned  u32x4;
typedef __attribute__((ext_vector_type(4)))  int       v4i;

constexpr int kB    = 4;
constexpr int kT    = 2048;
constexpr int kC    = 1024;
constexpr int kD    = 64;
constexpr float kEPS = 1e-6f;

#define AS1 __attribute__((address_space(1)))
#define AS3 __attribute__((address_space(3)))

#if defined(__has_builtin)
#if __has_builtin(__builtin_amdgcn_global_load_async_to_lds_b128)
#define HAVE_ASYNC_LDS 1
#endif
#endif
#ifndef HAVE_ASYNC_LDS
#define HAVE_ASYNC_LDS 0
#endif

static __device__ __forceinline__ void wait_async_then_barrier() {
#if HAVE_ASYNC_LDS
#if __has_builtin(__builtin_amdgcn_s_wait_asynccnt)
  __builtin_amdgcn_s_wait_asynccnt(0);
#else
  asm volatile("s_wait_asynccnt 0" ::: "memory");
#endif
#endif
  __syncthreads();
}

// ---------------------------------------------------------------------------
// Fragment loaders matching CDNA5 16-bit WMMA VGPR layouts (wave32).
// A (16xK row-major, lda elems): lane<16 -> row=lane, K=[0..7]+[16..23];
//                                lane>=16 -> row=lane-16, K=[8..15]+[24..31].
// B from Bt (NxK row-major):     lane<16 -> col=lane, K=[0..7]+[16..23];
//                                lane>=16 -> col=lane-16, K=[8..15]+[24..31].
// ---------------------------------------------------------------------------
static __device__ __forceinline__ v16bf load_a_frag(const __bf16* base, int lda) {
  const int lane = threadIdx.x & 31;
  const __bf16* p = base + (lane & 15) * lda + ((lane >> 4) << 3);
  union { v16bf v; u32x4 q[2]; } f;
  f.q[0] = *(const u32x4*)(p);
  f.q[1] = *(const u32x4*)(p + 16);
  return f.v;
}
static __device__ __forceinline__ v16bf load_b_frag(const __bf16* baseT, int ldb) {
  const int lane = threadIdx.x & 31;
  const __bf16* p = baseT + (lane & 15) * ldb + ((lane >> 4) << 3);
  union { v16bf v; u32x4 q[2]; } f;
  f.q[0] = *(const u32x4*)(p);
  f.q[1] = *(const u32x4*)(p + 16);
  return f.v;
}

// ---------------------------------------------------------------------------
// Conversion helpers
// ---------------------------------------------------------------------------
__global__ __launch_bounds__(256)
void cvt_f32_bf16_kernel(const float* __restrict__ in, __bf16* __restrict__ out, int n) {
  int i = blockIdx.x * 256 + threadIdx.x;
  if (i < n) out[i] = (__bf16)in[i];
}

// out[n*rowsK + k] = in[k*colsN + n]   (W^T, bf16)
__global__ __launch_bounds__(256)
void transpose_cvt_kernel(const float* __restrict__ in, __bf16* __restrict__ out,
                          int rowsK, int colsN) {
  int i = blockIdx.x * 256 + threadIdx.x;
  if (i < rowsK * colsN) {
    int n  = i / rowsK;
    int kk = i - n * rowsK;
    out[i] = (__bf16)in[(size_t)kk * colsN + n];
  }
}

// ---------------------------------------------------------------------------
// Tiled WMMA GEMM: C[m,n] = A[m,:] . Bt[n,:] + bias[n]
// Block tile 128x128, 8 waves, each wave 32x64 (2x4 WMMA tiles), K-step 32.
// Staging: async global->LDS (ASYNCcnt) if available, else b128 load/ds_store.
// mode 0: qkv epilogue (elu+1 on q,k; scatter to (b,h,t,d) bf16 buffers)
// mode 1: proj epilogue (f32 store to out[m*N+n])
// ---------------------------------------------------------------------------
__global__ __launch_bounds__(256)
void gemm_bf16_kernel(const __bf16* __restrict__ A, const __bf16* __restrict__ Bt,
                      const float* __restrict__ bias, int K, int N, int mode,
                      float* __restrict__ outF, __bf16* __restrict__ qb,
                      __bf16* __restrict__ kb, __bf16* __restrict__ vb) {
  __shared__ __align__(16) __bf16 lA[128][40];
  __shared__ __align__(16) __bf16 lB[128][40];

  const int tid  = threadIdx.x;
  const int lane = tid & 31;
  const int w    = tid >> 5;
  const int rm   = (w >> 1) * 32;   // wave row base in tile
  const int cn   = (w & 1) * 64;    // wave col base in tile
  const int m0   = blockIdx.x * 128;
  const int n0   = blockIdx.y * 128;

  const v8f vzero = {0.f, 0.f, 0.f, 0.f, 0.f, 0.f, 0.f, 0.f};
  v8f acc[2][4];
#pragma unroll
  for (int i = 0; i < 2; ++i)
#pragma unroll
    for (int j = 0; j < 4; ++j) acc[i][j] = vzero;

  for (int k0 = 0; k0 < K; k0 += 32) {
    __syncthreads();
#pragma unroll
    for (int it = 0; it < 2; ++it) {
      int idx = tid + it * 256;           // 512 chunks of 8 halves
      int r = idx >> 2;
      int c = (idx & 3) << 3;
      const __bf16* ga = A  + (size_t)(m0 + r) * K + k0 + c;
      const __bf16* gb = Bt + (size_t)(n0 + r) * K + k0 + c;
#if HAVE_ASYNC_LDS
      __builtin_amdgcn_global_load_async_to_lds_b128(
          (AS1 v4i*)(__bf16*)ga, (AS3 v4i*)&lA[r][c], 0, 0);
      __builtin_amdgcn_global_load_async_to_lds_b128(
          (AS1 v4i*)(__bf16*)gb, (AS3 v4i*)&lB[r][c], 0, 0);
#else
      *(u32x4*)&lA[r][c] = *(const u32x4*)(ga);
      *(u32x4*)&lB[r][c] = *(const u32x4*)(gb);
      __builtin_prefetch(ga + 32, 0, 0);  // speculative: next K tile
      __builtin_prefetch(gb + 32, 0, 0);
#endif
    }
    wait_async_then_barrier();

    v16bf af[2], bfr[4];
#pragma unroll
    for (int i = 0; i < 2; ++i) af[i] = load_a_frag(&lA[rm + i * 16][0], 40);
#pragma unroll
    for (int j = 0; j < 4; ++j) bfr[j] = load_b_frag(&lB[cn + j * 16][0], 40);
#pragma unroll
    for (int i = 0; i < 2; ++i)
#pragma unroll
      for (int j = 0; j < 4; ++j)
        acc[i][j] = __builtin_amdgcn_wmma_f32_16x16x32_bf16(
            false, af[i], false, bfr[j], (short)0, acc[i][j], false, false);
  }

  // Epilogue.  C/D layout: vgpr r, lane<16 -> (m=r,n=lane); lane>=16 -> (m=r+8,n=lane-16)
  const int hi8 = (lane >> 4) << 3;
  const int nl  = lane & 15;
#pragma unroll
  for (int i = 0; i < 2; ++i) {
#pragma unroll
    for (int j = 0; j < 4; ++j) {
#pragma unroll
      for (int r = 0; r < 8; ++r) {
        int m = m0 + rm + i * 16 + r + hi8;
        int n = n0 + cn + j * 16 + nl;
        float val = acc[i][j][r] + bias[n];
        if (mode == 0) {
          int which = n >> 10;        // 0:q 1:k 2:v
          int cch   = n & 1023;
          int h     = cch >> 6;
          int dd    = cch & 63;
          int b     = m >> 11;        // T = 2048
          int t     = m & 2047;
          if (which < 2) val = (val > 0.f) ? (val + 1.f) : __expf(val);  // elu+1
          __bf16* dst = (which == 0) ? qb : ((which == 1) ? kb : vb);
          dst[((size_t)((b << 4) + h) * kT + t) * kD + dd] = (__bf16)val;
        } else {
          outF[(size_t)m * N + n] = val;
        }
      }
    }
  }
}

// ---------------------------------------------------------------------------
// Chunk-parallel causal linear attention (3 phases).
// S1: per (b,h,chunk): KVc[j][i] = sum_s V[s][j] K[s][i]  (bf16 out), plus
//     chunk column-sums of K (f32).  Fully parallel: 2048 blocks.
// ---------------------------------------------------------------------------
__global__ __launch_bounds__(128)
void chunk_kv_kernel(const __bf16* __restrict__ k, const __bf16* __restrict__ v,
                     __bf16* __restrict__ kvc, float* __restrict__ ksc) {
  __shared__ __align__(16) __bf16 sKt[64 * 64];   // [d][s]
  __shared__ __align__(16) __bf16 sVt[64 * 64];   // [d][s]
  const int tid  = threadIdx.x;
  const int lane = tid & 31;
  const int w    = tid >> 5;
  const int tm   = w * 16;
  const int bh   = blockIdx.x >> 5;
  const int c    = blockIdx.x & 31;
  const int t0   = c * 64;
  const __bf16* kh = k + (size_t)bh * kT * kD;
  const __bf16* vh = v + (size_t)bh * kT * kD;

  for (int i = tid; i < 64 * 64; i += 128) {
    int s = i >> 6, d = i & 63;
    sKt[d * 64 + s] = kh[(size_t)(t0 + s) * kD + d];
    sVt[d * 64 + s] = vh[(size_t)(t0 + s) * kD + d];
  }
  __syncthreads();

  const int hi8 = (lane >> 4) << 3;
  const int nl  = lane & 15;
  const v8f vzero = {0.f, 0.f, 0.f, 0.f, 0.f, 0.f, 0.f, 0.f};
  __bf16* out = kvc + (size_t)blockIdx.x * 4096;

  v16bf aV0 = load_a_frag(&sVt[tm * 64], 64);
  v16bf aV1 = load_a_frag(&sVt[tm * 64 + 32], 64);
#pragma unroll
  for (int nt = 0; nt < 4; ++nt) {
    v16bf bK0 = load_b_frag(&sKt[nt * 16 * 64], 64);
    v16bf bK1 = load_b_frag(&sKt[nt * 16 * 64 + 32], 64);
    v8f u = vzero;
    u = __builtin_amdgcn_wmma_f32_16x16x32_bf16(false, aV0, false, bK0, (short)0, u, false, false);
    u = __builtin_amdgcn_wmma_f32_16x16x32_bf16(false, aV1, false, bK1, (short)0, u, false, false);
#pragma unroll
    for (int r = 0; r < 8; ++r) {
      int m = tm + r + hi8;             // j
      int n = nt * 16 + nl;             // i
      out[m * 64 + n] = (__bf16)u[r];
    }
  }
  if (tid < 64) {
    float s = 0.f;
    for (int ss = 0; ss < 64; ++ss) s += (float)sKt[tid * 64 + ss];
    ksc[(size_t)blockIdx.x * 64 + tid] = s;
  }
}

// ---------------------------------------------------------------------------
// S2: in-place exclusive prefix over the 32 chunks per (b,h): KV state before
// each chunk (bf16) and ksum before each chunk (f32).  64 blocks, tiny.
// ---------------------------------------------------------------------------
__global__ __launch_bounds__(128)
void prefix_kv_kernel(__bf16* __restrict__ kvc, float* __restrict__ ksc) {
  const int bh = blockIdx.x;
  for (int e = threadIdx.x; e < 4096; e += 128) {
    size_t base = (size_t)bh * 32 * 4096 + e;
    float acc = 0.f;
    for (int c = 0; c < 32; ++c) {
      float tmp = (float)kvc[base + (size_t)c * 4096];
      kvc[base + (size_t)c * 4096] = (__bf16)acc;
      acc += tmp;
    }
  }
  for (int e = threadIdx.x; e < 64; e += 128) {
    size_t base = (size_t)bh * 32 * 64 + e;
    float acc = 0.f;
    for (int c = 0; c < 32; ++c) {
      float tmp = ksc[base + (size_t)c * 64];
      ksc[base + (size_t)c * 64] = acc;
      acc += tmp;
    }
  }
}

// ---------------------------------------------------------------------------
// S3: per (b,h,chunk): S = causal_mask(Q K^T); den = rowsum(S) + Q.ksum + eps;
//     Y = (S V + Q KVprefix) / den.  Fully parallel: 2048 blocks.
// ---------------------------------------------------------------------------
__global__ __launch_bounds__(128)
void chunk_out_kernel(const __bf16* __restrict__ q, const __bf16* __restrict__ k,
                      const __bf16* __restrict__ v, const __bf16* __restrict__ kvp,
                      const float* __restrict__ ksp, __bf16* __restrict__ yb) {
  __shared__ __align__(16) __bf16 sVt[64 * 64];   // [d][s]
  __shared__ __align__(16) __bf16 sS[64 * 64];    // masked S [t][s]
  __shared__ float sden[64];

  const int tid  = threadIdx.x;
  const int lane = tid & 31;
  const int w    = tid >> 5;
  const int tm   = w * 16;
  const int bh   = blockIdx.x >> 5;
  const int c    = blockIdx.x & 31;
  const int b    = bh >> 4;
  const int h    = bh & 15;
  const int t0   = c * 64;
  const __bf16* qh = q + (size_t)bh * kT * kD;
  const __bf16* kh = k + (size_t)bh * kT * kD;
  const __bf16* vh = v + (size_t)bh * kT * kD;
  const __bf16* kvb = kvp + (size_t)blockIdx.x * 4096;  // KV prefix [j][i]
  const float*  ks  = ksp + (size_t)blockIdx.x * 64;

  for (int i = tid; i < 64 * 64; i += 128) {
    int s = i >> 6, d = i & 63;
    sVt[d * 64 + s] = vh[(size_t)(t0 + s) * kD + d];
  }

  const int hi8 = (lane >> 4) << 3;
  const int nl  = lane & 15;
  const v8f vzero = {0.f, 0.f, 0.f, 0.f, 0.f, 0.f, 0.f, 0.f};

  v16bf aQ0 = load_a_frag(qh + (size_t)(t0 + tm) * kD, kD);
  v16bf aQ1 = load_a_frag(qh + (size_t)(t0 + tm) * kD + 32, kD);

  // S = causal_mask(Q K^T)   (Bt = K row-major, direct from global)
#pragma unroll
  for (int nt = 0; nt < 4; ++nt) {
    v16bf b0 = load_b_frag(kh + (size_t)(t0 + nt * 16) * kD, kD);
    v16bf b1 = load_b_frag(kh + (size_t)(t0 + nt * 16) * kD + 32, kD);
    v8f s_acc = vzero;
    s_acc = __builtin_amdgcn_wmma_f32_16x16x32_bf16(false, aQ0, false, b0, (short)0, s_acc, false, false);
    s_acc = __builtin_amdgcn_wmma_f32_16x16x32_bf16(false, aQ1, false, b1, (short)0, s_acc, false, false);
#pragma unroll
    for (int r = 0; r < 8; ++r) {
      int m = tm + r + hi8;
      int n = nt * 16 + nl;
      sS[m * 64 + n] = (__bf16)((n <= m) ? s_acc[r] : 0.f);
    }
  }
  __syncthreads();

  // denominators
  if (tid < 64) {
    int t = tid;
    float dsum = 0.f;
    for (int s = 0; s < 64; ++s) dsum += (float)sS[t * 64 + s];
    const __bf16* qrow = qh + (size_t)(t0 + t) * kD;
    for (int i = 0; i < 64; ++i) dsum += (float)qrow[i] * ks[i];
    sden[t] = dsum + kEPS;
  }
  __syncthreads();

  // Y = (S V + Q KVp) / den
  v16bf aS0 = load_a_frag(&sS[tm * 64], 64);
  v16bf aS1 = load_a_frag(&sS[tm * 64 + 32], 64);
#pragma unroll
  for (int nt = 0; nt < 4; ++nt) {
    v16bf bV0 = load_b_frag(&sVt[nt * 16 * 64], 64);
    v16bf bV1 = load_b_frag(&sVt[nt * 16 * 64 + 32], 64);
    v16bf bK0 = load_b_frag(kvb + nt * 16 * 64, 64);
    v16bf bK1 = load_b_frag(kvb + nt * 16 * 64 + 32, 64);
    v8f y_acc = vzero;
    y_acc = __builtin_amdgcn_wmma_f32_16x16x32_bf16(false, aS0, false, bV0, (short)0, y_acc, false, false);
    y_acc = __builtin_amdgcn_wmma_f32_16x16x32_bf16(false, aS1, false, bV1, (short)0, y_acc, false, false);
    y_acc = __builtin_amdgcn_wmma_f32_16x16x32_bf16(false, aQ0, false, bK0, (short)0, y_acc, false, false);
    y_acc = __builtin_amdgcn_wmma_f32_16x16x32_bf16(false, aQ1, false, bK1, (short)0, y_acc, false, false);
#pragma unroll
    for (int r = 0; r < 8; ++r) {
      int m = tm + r + hi8;
      int n = nt * 16 + nl;
      float yv = y_acc[r] / sden[m];
      yb[((size_t)(b * kT + t0 + m)) * kC + h * kD + n] = (__bf16)yv;
    }
  }
}

// ---------------------------------------------------------------------------
// Host launcher
// ---------------------------------------------------------------------------
extern "C" void kernel_launch(void* const* d_in, const int* in_sizes, int n_in,
                              void* d_out, int out_size, void* d_ws, size_t ws_size,
                              hipStream_t stream) {
  (void)in_sizes; (void)n_in; (void)out_size; (void)ws_size;
  const float* x      = (const float*)d_in[0];
  const float* W_attn = (const float*)d_in[1];
  const float* b_attn = (const float*)d_in[2];
  const float* W_proj = (const float*)d_in[3];
  const float* b_proj = (const float*)d_in[4];
  float* out = (float*)d_out;

  char* ws = (char*)d_ws;
  size_t off = 0;
  auto alloc = [&](size_t bytes) -> void* {
    void* p = ws + off;
    off += (bytes + 255) & ~(size_t)255;
    return p;
  };
  const size_t nX   = (size_t)kB * kT * kC;       // 8388608
  const size_t nQKV = (size_t)kB * 16 * kT * kD;  // 8388608 per tensor
  __bf16* xb  = (__bf16*)alloc(nX * 2);
  __bf16* WaT = (__bf16*)alloc((size_t)3 * kC * kC * 2);
  __bf16* WpT = (__bf16*)alloc((size_t)kC * kC * 2);
  __bf16* qf  = (__bf16*)alloc(nQKV * 2);
  __bf16* kf  = (__bf16*)alloc(nQKV * 2);
  __bf16* vf  = (__bf16*)alloc(nQKV * 2);
  __bf16* yb  = (__bf16*)alloc(nX * 2);
  __bf16* kvc = (__bf16*)alloc((size_t)64 * 32 * 4096 * 2);  // chunk KV / prefix
  float*  ksc = (float*) alloc((size_t)64 * 32 * 64 * 4);    // chunk ksum / prefix

  cvt_f32_bf16_kernel<<<(int)((nX + 255) / 256), 256, 0, stream>>>(x, xb, (int)nX);
  transpose_cvt_kernel<<<(3 * kC * kC + 255) / 256, 256, 0, stream>>>(W_attn, WaT, kC, 3 * kC);
  transpose_cvt_kernel<<<(kC * kC + 255) / 256, 256, 0, stream>>>(W_proj, WpT, kC, kC);

  // qkv = x @ W_attn + b ; fused elu+1 on q,k ; scatter to (b,h,t,d)
  gemm_bf16_kernel<<<dim3(64, 24), 256, 0, stream>>>(
      xb, WaT, b_attn, kC, 3 * kC, 0, nullptr, qf, kf, vf);

  // chunk-parallel causal linear attention
  chunk_kv_kernel<<<64 * 32, 128, 0, stream>>>(kf, vf, kvc, ksc);
  prefix_kv_kernel<<<64, 128, 0, stream>>>(kvc, ksc);
  chunk_out_kernel<<<64 * 32, 128, 0, stream>>>(qf, kf, vf, kvc, ksc, yb);

  // out = y @ W_proj + b_proj
  gemm_bf16_kernel<<<dim3(64, 8), 256, 0, stream>>>(
      yb, WpT, b_proj, kC, kC, 1, out, nullptr, nullptr, nullptr);
}
